// TextRNN_55602646614590
// MI455X (gfx1250) — matrix operand order repaired and checked by reading, not compiled, not run on hardware
//
#include <hip/hip_runtime.h>
#include <stdint.h>
#include <math.h>

// Problem dims (fixed by the reference)
#define BB 64
#define SS 256
#define VV 32000
#define EE 1024
#define HH 1024

typedef __attribute__((ext_vector_type(16))) __bf16 v16bf;
typedef __attribute__((ext_vector_type(8)))  __bf16 v8bf;
typedef __attribute__((ext_vector_type(8)))  float  v8f;
typedef __attribute__((ext_vector_type(4)))  uint32_t v4u;
typedef __attribute__((ext_vector_type(8)))  uint32_t v8u;

static __device__ __forceinline__ __bf16 f2bf(float f) {
  uint32_t u = __builtin_bit_cast(uint32_t, f);
  u += 0x7FFFu + ((u >> 16) & 1u);          // round-to-nearest-even
  uint16_t h = (uint16_t)(u >> 16);
  return __builtin_bit_cast(__bf16, h);
}

// ---------------------------------------------------------------------------
// Embedding gather + fp32->bf16; output rows ordered m = s*B + b so the xproj
// GEMM result lands directly in [S][B][4][H] layout.
// ---------------------------------------------------------------------------
__global__ __launch_bounds__(256) void embed_kernel(const int* __restrict__ X,
                                                    const float* __restrict__ C,
                                                    __bf16* __restrict__ emb) {
  int bid = blockIdx.x;                 // = b*S + s
  int b = bid / SS, s = bid % SS;
  int tok = X[bid];
  const float* src = C + (size_t)tok * EE;
  __bf16* dst = emb + ((size_t)s * BB + b) * EE;
  int e = threadIdx.x * 4;              // E = 1024 = 256*4
  float4 v = *(const float4*)(src + e);
  uint64_t p = (uint64_t)__builtin_bit_cast(uint16_t, f2bf(v.x))
             | ((uint64_t)__builtin_bit_cast(uint16_t, f2bf(v.y)) << 16)
             | ((uint64_t)__builtin_bit_cast(uint16_t, f2bf(v.z)) << 32)
             | ((uint64_t)__builtin_bit_cast(uint16_t, f2bf(v.w)) << 48);
  *(uint64_t*)(dst + e) = p;
}

// ---------------------------------------------------------------------------
// Transpose + convert: src is G stacked K x N fp32 matrices (row-major),
// dst is G stacked N x K bf16 matrices (i.e. B^T, the layout the GEMM's B loads use:
// 16 contiguous K-values per output column).
// ---------------------------------------------------------------------------
__global__ __launch_bounds__(256) void transpose_bf16_kernel(const float* __restrict__ src,
                                                             __bf16* __restrict__ dst,
                                                             int K, int N, long long total) {
  long long idx = (long long)blockIdx.x * 256 + threadIdx.x;
  if (idx >= total) return;
  long long kn = (long long)K * N;
  long long g = idx / kn;
  long long rem = idx - g * kn;
  int k = (int)(rem / N);
  int n = (int)(rem - (long long)k * N);
  dst[(g * N + n) * (long long)K + k] = f2bf(src[idx]);
}

// ---------------------------------------------------------------------------
// Init LSTM state (h=0, c=0) and fold the three gate biases: bx + bh + bg.
// ---------------------------------------------------------------------------
__global__ __launch_bounds__(256) void init_state_kernel(const float* __restrict__ bx,
                                                         const float* __restrict__ bh,
                                                         const float* __restrict__ bg,
                                                         float* __restrict__ bias_total,
                                                         float* __restrict__ cbuf,
                                                         __bf16* __restrict__ hbf) {
  int idx = blockIdx.x * 256 + threadIdx.x;     // 0 .. B*H-1 (65536)
  cbuf[idx] = 0.0f;
  hbf[idx] = f2bf(0.0f);
  if (idx < 4 * HH) bias_total[idx] = bx[idx] + bh[idx] + bg[idx];
}

// ---------------------------------------------------------------------------
// Generic bf16 WMMA GEMM:  out[M][ldOut] = A[M][K] * Bt[N][K]^T (+addend)(+bias)
// Block = 256 threads = 8 waves, block tile 32(M) x 256(N).
// Wave (w): rows (w/4)*16, cols (w%4)*64 -> 4 WMMA tiles, K-loop step 32.
// A tile (32x32 bf16, 2KB) staged in LDS via the Tensor Data Mover
// (tensor_load_to_lds, TENSORcnt); B fragments streamed from global
// (contiguous 32B per lane thanks to the Bt layout).
// ---------------------------------------------------------------------------
__global__ __launch_bounds__(256) void gemm_bf16_wmma_kernel(
    const __bf16* __restrict__ A,      // [M][K] row-major, M = gridDim.x*32
    const __bf16* __restrict__ Bt,     // [N][K] row-major (transposed B)
    float* __restrict__ out,           // [M][ldOut]
    int K, int ldOut,
    const float* __restrict__ addend, int ldAdd,   // optional, [M][ldAdd]
    const float* __restrict__ bias)                // optional, [N]
{
  __shared__ __bf16 Atile[32][32];

  const int tid   = threadIdx.x;
  const int lane  = tid & 31;          // wave32
  const int wave  = tid >> 5;
  const int half  = lane >> 4;         // 0 | 1
  const int l16   = lane & 15;
  const int mBlk  = blockIdx.x * 32;
  const int mWave = (wave >> 2) * 16;
  const int nBase = blockIdx.y * 256 + (wave & 3) * 64;

  v8f acc[4] = {};

  // TDM descriptor invariants for this block: 32 rows x 32 K-elems of A,
  // row stride K elements, bf16 (data_size=1 -> 2 bytes), dest = Atile.
  const uint32_t ldsAddr = (uint32_t)(uintptr_t)(&Atile[0][0]);
  const uint64_t abase   = (uint64_t)(uintptr_t)(A + (size_t)mBlk * K);

  for (int k0 = 0; k0 < K; k0 += 32) {
    if (wave == 0) {
      const uint64_t gaddr = abase + (uint64_t)k0 * 2;   // byte address of tile
      v4u g0;
      g0[0] = 1u;                                        // count=1 (valid user D#)
      g0[1] = ldsAddr;                                   // lds_addr [63:32]
      g0[2] = (uint32_t)gaddr;                           // global_addr lo [95:64]
      g0[3] = (uint32_t)((gaddr >> 32) & 0x01FFFFFFu)    // global_addr hi [120:96]
            | (2u << 30);                                // type=2 ("image") [127:126]
      v8u g1;
      g1[0] = 0x00010000u;                  // workgroup_mask=0 | data_size=1 (2B)
      g1[1] = (uint32_t)(K & 0xFFFF) << 16; // tensor_dim0[15:0] @ bits[63:48]
      g1[2] = (uint32_t)((K >> 16) & 0xFFFF)// tensor_dim0[31:16]
            | (32u << 16);                  // tensor_dim1 = 32 rows
      g1[3] = (32u << 16);                  // tensor_dim1 hi=0 | tile_dim0=32
      g1[4] = 32u;                          // tile_dim1=32 | tile_dim2=0
      g1[5] = (uint32_t)K;                  // tensor_dim0_stride lo32 (elements)
      g1[6] = 0u;                           // stride0 hi | stride1 lo
      g1[7] = 0u;                           // stride1 hi
      asm volatile("tensor_load_to_lds %0, %1" :: "s"(g0), "s"(g1) : "memory");
      __builtin_amdgcn_s_wait_tensorcnt(0);
    }
    __syncthreads();

    // A fragment, contiguous K order (matches B): lanes 0-15 -> K {0..15}, 16-31 -> {16..31}
    const __bf16* ap = &Atile[mWave + l16][0];
    v8bf lo = *(const v8bf*)(ap + half * 16);
    v8bf hi = *(const v8bf*)(ap + half * 16 + 8);
    v16bf a;
#pragma unroll
    for (int j = 0; j < 8; ++j) { a[j] = lo[j]; a[j + 8] = hi[j]; }

#pragma unroll
    for (int t = 0; t < 4; ++t) {
      // B fragment: lane = column, 16 contiguous K values starting at half*16
      const __bf16* bp = Bt + (size_t)(nBase + t * 16 + l16) * K + k0 + half * 16;
      __builtin_prefetch(bp + 32, 0, 3);     // speculative, next K-step
      v16bf b = *(const v16bf*)bp;
      acc[t] = __builtin_amdgcn_wmma_f32_16x16x32_bf16(
          /*neg_a=*/false, a, /*neg_b=*/false, b,
          /*c_mod=*/(short)0, acc[t], /*reuse_a=*/false, /*reuse_b=*/false);
    }
    __syncthreads();
  }

  // C/D layout: VGPR i -> M = i + 8*half, N = lane&15
  const int rowW = mBlk + mWave + half * 8;
#pragma unroll
  for (int t = 0; t < 4; ++t) {
    const int n = nBase + t * 16 + l16;
    const float badd = bias ? bias[n] : 0.0f;
#pragma unroll
    for (int i = 0; i < 8; ++i) {
      const int row = rowW + i;
      float v = acc[t][i] + badd;
      if (addend) v += addend[(size_t)row * ldAdd + n];
      out[(size_t)row * ldOut + n] = v;
    }
  }
}

// ---------------------------------------------------------------------------
// LSTM pointwise gates: gbuf[b][4*H] already = xproj[s] + h@Wh + (bx+bh+bg)
// ---------------------------------------------------------------------------
__global__ __launch_bounds__(256) void lstm_gate_kernel(const float* __restrict__ gbuf,
                                                        float* __restrict__ cbuf,
                                                        __bf16* __restrict__ hbf) {
  int idx = blockIdx.x * 256 + threadIdx.x;   // 0 .. B*H-1
  int b = idx >> 10;
  int h = idx & (HH - 1);
  const float* g = gbuf + (size_t)b * (4 * HH);
  float gi = g[h];
  float gf = g[HH + h];
  float go = g[2 * HH + h];
  float gc = g[3 * HH + h];
  float si = 1.0f / (1.0f + __expf(-gi));
  float sf = 1.0f / (1.0f + __expf(-gf));
  float so = 1.0f / (1.0f + __expf(-go));
  float c = sf * cbuf[idx] + si * tanhf(gc);
  cbuf[idx] = c;
  hbf[idx] = f2bf(so * tanhf(c));
}

// ---------------------------------------------------------------------------
// Host driver. Workspace usage ~386 MB:
//   emb_bf 33.5M | WxT 8.4M | WhT 8.4M | WoutT 65.5M | xproj 268.4M
//   gbuf 1M | cbuf 0.25M | hbf 0.13M | bias_total 16K
// ---------------------------------------------------------------------------
extern "C" void kernel_launch(void* const* d_in, const int* in_sizes, int n_in,
                              void* d_out, int out_size, void* d_ws, size_t ws_size,
                              hipStream_t stream) {
  const int*   X    = (const int*)d_in[0];
  const float* C    = (const float*)d_in[1];
  const float* Wx   = (const float*)d_in[2];
  const float* bx   = (const float*)d_in[3];
  const float* Wh   = (const float*)d_in[4];
  const float* bh   = (const float*)d_in[5];
  const float* bg   = (const float*)d_in[6];
  const float* Wout = (const float*)d_in[7];
  const float* bout = (const float*)d_in[8];
  float* out = (float*)d_out;

  char* ws = (char*)d_ws;
  size_t off = 0;
  auto carve = [&](size_t bytes) -> void* {
    void* p = ws + off;
    off = (off + bytes + 255) & ~(size_t)255;
    return p;
  };

  __bf16* emb   = (__bf16*)carve((size_t)BB * SS * EE * 2);
  __bf16* WxT   = (__bf16*)carve((size_t)4 * HH * EE * 2);
  __bf16* WhT   = (__bf16*)carve((size_t)4 * HH * HH * 2);
  __bf16* WoT   = (__bf16*)carve((size_t)VV * HH * 2);
  float*  xproj = (float*)carve((size_t)BB * SS * 4 * HH * 4);
  float*  gbuf  = (float*)carve((size_t)BB * 4 * HH * 4);
  float*  cbuf  = (float*)carve((size_t)BB * HH * 4);
  __bf16* hbf   = (__bf16*)carve((size_t)BB * HH * 2);
  float*  btot  = (float*)carve((size_t)4 * HH * 4);

  // 1) embed + convert
  embed_kernel<<<BB * SS, 256, 0, stream>>>(X, C, emb);

  // 2) weight transposes to bf16 B^T layouts
  { long long tot = 4LL * EE * HH;
    transpose_bf16_kernel<<<(unsigned)((tot + 255) / 256), 256, 0, stream>>>(Wx, WxT, EE, HH, tot); }
  { long long tot = 4LL * HH * HH;
    transpose_bf16_kernel<<<(unsigned)((tot + 255) / 256), 256, 0, stream>>>(Wh, WhT, HH, HH, tot); }
  { long long tot = 1LL * HH * VV;
    transpose_bf16_kernel<<<(unsigned)((tot + 255) / 256), 256, 0, stream>>>(Wout, WoT, HH, VV, tot); }

  // 3) state init + fused bias
  init_state_kernel<<<(BB * HH) / 256, 256, 0, stream>>>(bx, bh, bg, btot, cbuf, hbf);

  // 4) xproj = emb @ Wx + (bx+bh+bg), M=16384 N=4096 K=1024
  gemm_bf16_wmma_kernel<<<dim3(BB * SS / 32, 4 * HH / 256), 256, 0, stream>>>(
      emb, WxT, xproj, EE, 4 * HH, nullptr, 0, btot);

  // 5) 256 sequential LSTM steps: g = h@Wh + xproj[s]; gates pointwise
  for (int s = 0; s < SS; ++s) {
    gemm_bf16_wmma_kernel<<<dim3(BB / 32, 4 * HH / 256), 256, 0, stream>>>(
        hbf, WhT, gbuf, HH, 4 * HH, xproj + (size_t)s * BB * 4 * HH, 4 * HH, nullptr);
    lstm_gate_kernel<<<(BB * HH) / 256, 256, 0, stream>>>(gbuf, cbuf, hbf);
  }

  // 6) classifier: out = h_last @ Wout + bout, M=64 N=32000 K=1024
  gemm_bf16_wmma_kernel<<<dim3(BB / 32, VV / 256), 256, 0, stream>>>(
      hbf, WoT, out, HH, VV, nullptr, 0, bout);
}
